// OPTAttention_22419729285477
// MI455X (gfx1250) — matrix operand order repaired and checked by reading, not compiled, not run on hardware
//
#include <hip/hip_runtime.h>

typedef _Float16 v16h __attribute__((ext_vector_type(16)));
typedef _Float16 v8h  __attribute__((ext_vector_type(8)));
typedef _Float16 v4h  __attribute__((ext_vector_type(4)));
typedef _Float16 v2h  __attribute__((ext_vector_type(2)));
typedef float    v8f  __attribute__((ext_vector_type(8)));
typedef float    v4f  __attribute__((ext_vector_type(4)));

#define DMODEL 2048
#define NHEAD  32
#define HDIM   64
#define SEQ    2048
#define BATCH  2

__device__ __forceinline__ v8f wmma_f16(v16h a, v16h b, v8f c) {
  // D = A(16x32 f16) x B(32x16 f16) + C(16x16 f32)
  return __builtin_amdgcn_wmma_f32_16x16x32_f16(false, a, false, b, (short)0, c,
                                                false, false);
}

__device__ __forceinline__ v8f zero8() {
  v8f z = {0.f, 0.f, 0.f, 0.f, 0.f, 0.f, 0.f, 0.f};
  return z;
}

// CDNA5 async memory->LDS copy (ASYNCcnt-tracked, no VGPR round trip).
// LDS address comes from the vdst VGPR; 16B chunks.
__device__ __forceinline__ void async_copy_b128(unsigned lds_off,
                                                const void* gptr) {
  asm volatile("global_load_async_to_lds_b128 %0, %1, off"
               :
               : "v"(lds_off), "v"((unsigned long long)(uintptr_t)gptr)
               : "memory");
}
__device__ __forceinline__ void wait_async0() {
  asm volatile("s_wait_asynccnt 0" ::: "memory");
}
__device__ __forceinline__ unsigned lds_off32(const void* p) {
  return (unsigned)(uintptr_t)p;  // flat addr[31:0] == LDS offset (ISA 10.2)
}

union H16 { v16h v; v2h p[8]; };

// A-fragment gather from an LDS tile stored [m][k] (row-major, leading dim ld).
// ISA 16-bit A layout: lane L (m = L%16), VGPR v<4 -> k = 2v + 8*(L/16),
// v>=4 -> k = 16 + 2(v-4) + 8*(L/16). Pairs (k,k+1) contiguous -> b32 loads.
__device__ __forceinline__ v16h load_afrag(const _Float16* t, int row, int ld,
                                           int k0, int lane) {
  const int hb = (lane >> 4) << 3;
  const _Float16* p = t + row * ld + k0 + hb;
  H16 f;
#pragma unroll
  for (int v = 0; v < 4; ++v) f.p[v] = *(const v2h*)(p + 2 * v);
#pragma unroll
  for (int v = 0; v < 4; ++v) f.p[4 + v] = *(const v2h*)(p + 16 + 2 * v);
  return f.v;
}

// B-fragment gather from an LDS tile stored [n][k] (leading dim ld).
// ISA 16-bit B layout: lane L (n = L%16), VGPR v<4 -> k = 2v + 8*(L/16),
// v>=4 -> k = 16 + 2(v-4) + 8*(L/16).
__device__ __forceinline__ v16h load_bfrag(const _Float16* t, int col, int ld,
                                           int k0, int lane) {
  const int hb = (lane >> 4) << 3;
  const _Float16* p = t + col * ld + k0 + hb;
  H16 f;
#pragma unroll
  for (int v = 0; v < 4; ++v) f.p[v] = *(const v2h*)(p + 2 * v);
#pragma unroll
  for (int v = 0; v < 4; ++v) f.p[4 + v] = *(const v2h*)(p + 16 + 2 * v);
  return f.v;
}

// ---------------------------------------------------------------------------
// f32 -> f16 conversion (vectorized x4)
// ---------------------------------------------------------------------------
__global__ __launch_bounds__(256) void cvt_f32_f16(const float* __restrict__ src,
                                                   _Float16* __restrict__ dst,
                                                   int n4) {
  int i = blockIdx.x * 256 + threadIdx.x;
  if (i >= n4) return;
  v4f x = *(const v4f*)(src + (size_t)i * 4);
  v4h y;
  y.x = (_Float16)x.x; y.y = (_Float16)x.y;
  y.z = (_Float16)x.z; y.w = (_Float16)x.w;
  *(v4h*)(dst + (size_t)i * 4) = y;
}

// ---------------------------------------------------------------------------
// GEMM: C[M=4096][N=2048] = A[4096][2048]_f16 * W[2048][2048]_f16 + bias
// Block tile 128x128, 8 waves (4x2), each wave 32x64 (8 WMMA accumulators).
// Double-buffered LDS; A-tile staged via async-to-LDS, W-tile transposed in
// registers (pair packing) since async copies cannot transpose.
// ---------------------------------------------------------------------------
template <bool F32OUT>
__global__ __launch_bounds__(256) void gemm128(const _Float16* __restrict__ A,
                                               const _Float16* __restrict__ W,
                                               const float* __restrict__ bias,
                                               void* __restrict__ Cout) {
  constexpr int LDA = 40;  // 128x32 tile, 80B row pitch (16B aligned)
  constexpr int LDB = 40;
  __shared__ _Float16 As[2][128 * LDA];
  __shared__ _Float16 Bs[2][128 * LDB];

  const int tid = threadIdx.x;
  const int lane = tid & 31, w = tid >> 5;
  const int wr = w >> 1, wc = w & 1;  // wave grid 4(m) x 2(n)
  const int mBase = blockIdx.y * 128, nBase = blockIdx.x * 128;
  const int nl = lane & 15, hf = lane >> 4;

  auto loadA = [&](int buf, int k0) {  // 512 b128 chunks, 2 per thread, async
#pragma unroll
    for (int i = 0; i < 2; ++i) {
      int idx = tid + i * 256;
      int row = idx >> 2, c8 = (idx & 3) * 8;
      async_copy_b128(lds_off32(&As[buf][row * LDA + c8]),
                      A + (size_t)(mBase + row) * DMODEL + k0 + c8);
    }
  };
  auto loadB = [&](int buf, int k0) {  // 32(k)x128(n) -> [n][k] pair packing
#pragma unroll
    for (int i = 0; i < 4; ++i) {
      int idx = tid + i * 256;           // 0..1023
      int np = idx & 63, kp = idx >> 6;  // n-pair, k-pair
      const _Float16* g = W + (size_t)(k0 + 2 * kp) * DMODEL + nBase + 2 * np;
      unsigned lo = *(const unsigned*)g;             // W[2kp][2np..2np+1]
      unsigned hi = *(const unsigned*)(g + DMODEL);  // W[2kp+1][2np..2np+1]
      unsigned s0 = (lo & 0xffffu) | (hi << 16);
      unsigned s1 = (lo >> 16) | (hi & 0xffff0000u);
      *(unsigned*)(&Bs[buf][(2 * np) * LDB + 2 * kp]) = s0;
      *(unsigned*)(&Bs[buf][(2 * np + 1) * LDB + 2 * kp]) = s1;
    }
  };

  v8f acc[2][4];
#pragma unroll
  for (int i = 0; i < 2; ++i)
#pragma unroll
    for (int j = 0; j < 4; ++j) acc[i][j] = zero8();

  loadA(0, 0);
  loadB(0, 0);
  wait_async0();
  __syncthreads();

  constexpr int NT = DMODEL / 32;  // 64 k-tiles
  for (int t = 0; t < NT; ++t) {
    const int cur = t & 1;
    if (t + 1 < NT) {  // prefetch next tile while computing current
      loadA(cur ^ 1, (t + 1) * 32);
      loadB(cur ^ 1, (t + 1) * 32);
    }
    v16h af0 = load_afrag(As[cur], wr * 32 + nl, LDA, 0, lane);
    v16h af1 = load_afrag(As[cur], wr * 32 + 16 + nl, LDA, 0, lane);
#pragma unroll
    for (int nt = 0; nt < 4; ++nt) {
      v16h bf = load_bfrag(Bs[cur], wc * 64 + nt * 16 + nl, LDB, 0, lane);
      acc[0][nt] = wmma_f16(af0, bf, acc[0][nt]);
      acc[1][nt] = wmma_f16(af1, bf, acc[1][nt]);
    }
    wait_async0();    // own async stores for next buffer landed
    __syncthreads();  // everyone done reading cur & writing cur^1
  }

  // Epilogue: C layout -> lane holds col n = L%16, rows r + 8*(L/16)
#pragma unroll
  for (int mt = 0; mt < 2; ++mt)
#pragma unroll
    for (int nt = 0; nt < 4; ++nt) {
      int n = nBase + wc * 64 + nt * 16 + nl;
      float bv = bias[n];
#pragma unroll
      for (int r = 0; r < 8; ++r) {
        int m = mBase + wr * 32 + mt * 16 + r + 8 * hf;
        float val = acc[mt][nt][r] + bv;
        if constexpr (F32OUT)
          ((float*)Cout)[(size_t)m * DMODEL + n] = val;
        else
          ((_Float16*)Cout)[(size_t)m * DMODEL + n] = (_Float16)val;
      }
    }
}

// ---------------------------------------------------------------------------
// Flash attention: one block per (b, h, 128-row q tile). 8 waves; each wave
// owns 16 query rows. kv tiles of 64. Causal + key-padding mask, online
// softmax. Q/K tiles staged via async-to-LDS; V transposed in registers.
// ---------------------------------------------------------------------------
__global__ __launch_bounds__(256) void flash64(const _Float16* __restrict__ Q,
                                               const _Float16* __restrict__ K,
                                               const _Float16* __restrict__ V,
                                               const int* __restrict__ amask,
                                               _Float16* __restrict__ ctx) {
  constexpr int LDQ = 72, LDK = 72;  // 144B row pitch: b128-aligned for async
  constexpr int LDV = 68, LDP = 68;
  __shared__ _Float16 Qs[128 * LDQ];  // [q][d]
  __shared__ _Float16 Ks[64 * LDK];   // [kv][d]
  __shared__ _Float16 Vt[64 * LDV];   // [d][kv]  (transposed)
  __shared__ _Float16 Ps[128 * LDP];  // [q][kv]  per-wave-private rows

  const int qt = blockIdx.x, h = blockIdx.y, b = blockIdx.z;
  const int tid = threadIdx.x, lane = tid & 31, w = tid >> 5;
  const int nl = lane & 15, hf = lane >> 4;
  const int qbase = qt * 128;
  const float inv_scale = 0.125f;  // 1/sqrt(64)
  const float NEG = -3.0e38f;      // ~mask_value

  // Q tile 128x64: 1024 async b128 units, 4 per thread
  {
    const _Float16* gq = Q + (size_t)(b * SEQ + qbase) * DMODEL + h * HDIM;
#pragma unroll
    for (int i = 0; i < 4; ++i) {
      int idx = tid + i * 256;
      int row = idx >> 3, c8 = (idx & 7) * 8;
      async_copy_b128(lds_off32(&Qs[row * LDQ + c8]),
                      gq + (size_t)row * DMODEL + c8);
    }
  }
  wait_async0();
  __syncthreads();

  v16h qf0 = load_afrag(Qs, w * 16 + nl, LDQ, 0, lane);
  v16h qf1 = load_afrag(Qs, w * 16 + nl, LDQ, 32, lane);

  v8f o[4];
  float mi[8], li[8];
#pragma unroll
  for (int d = 0; d < 4; ++d) o[d] = zero8();
#pragma unroll
  for (int r = 0; r < 8; ++r) { mi[r] = NEG; li[r] = 0.f; }

  const int ktiles = 2 * qt + 2;  // causal: kv <= qbase + 127
  for (int kt = 0; kt < ktiles; ++kt) {
    const int kvbase = kt * 64;
    __syncthreads();  // all waves finished reading previous Ks/Vt

    // K tile 64x64: 512 async b128 units, 2 per thread
    {
      const _Float16* gk = K + (size_t)(b * SEQ + kvbase) * DMODEL + h * HDIM;
#pragma unroll
      for (int i = 0; i < 2; ++i) {
        int idx = tid + i * 256;
        int row = idx >> 3, c8 = (idx & 7) * 8;
        async_copy_b128(lds_off32(&Ks[row * LDK + c8]),
                        gk + (size_t)row * DMODEL + c8);
      }
    }
    // V tile 64x64 -> Vt[d][kv] via 2x2 pair packing
    {
      const _Float16* gv = V + (size_t)(b * SEQ + kvbase) * DMODEL + h * HDIM;
#pragma unroll
      for (int i = 0; i < 4; ++i) {
        int idx = tid + i * 256;
        int np = idx & 31, kp = idx >> 5;  // d-pair, kv-pair
        const _Float16* g = gv + (size_t)(2 * kp) * DMODEL + 2 * np;
        unsigned lo = *(const unsigned*)g;
        unsigned hi = *(const unsigned*)(g + DMODEL);
        unsigned s0 = (lo & 0xffffu) | (hi << 16);
        unsigned s1 = (lo >> 16) | (hi & 0xffff0000u);
        *(unsigned*)(&Vt[(2 * np) * LDV + 2 * kp]) = s0;
        *(unsigned*)(&Vt[(2 * np + 1) * LDV + 2 * kp]) = s1;
      }
    }
    wait_async0();
    __syncthreads();

    // S = Q K^T : per-wave 16x64 (contraction over d=64; K tile is [n][k])
    v8f s[4];
#pragma unroll
    for (int nt = 0; nt < 4; ++nt) {
      s[nt] = zero8();
      v16h bf0 = load_bfrag(Ks, nt * 16 + nl, LDK, 0, lane);
      s[nt] = wmma_f16(qf0, bf0, s[nt]);
      v16h bf1 = load_bfrag(Ks, nt * 16 + nl, LDK, 32, lane);
      s[nt] = wmma_f16(qf1, bf1, s[nt]);
    }

    // Scale + causal/padding mask (C layout: n = nl, m = r + 8*hf)
    const bool diag = (kt >= 2 * qt);
#pragma unroll
    for (int nt = 0; nt < 4; ++nt) {
      int n_g = kvbase + nt * 16 + nl;
      int key_ok = amask[b * SEQ + n_g];
#pragma unroll
      for (int r = 0; r < 8; ++r) {
        float x = s[nt][r] * inv_scale;
        int m_g = qbase + w * 16 + r + 8 * hf;
        if ((diag && n_g > m_g) || key_ok == 0) x = NEG;
        s[nt][r] = x;
      }
    }

    // Online softmax update (row reductions across 16-lane half-waves)
#pragma unroll
    for (int r = 0; r < 8; ++r) {
      float v = fmaxf(fmaxf(s[0][r], s[1][r]), fmaxf(s[2][r], s[3][r]));
#pragma unroll
      for (int sh = 1; sh <= 8; sh <<= 1) v = fmaxf(v, __shfl_xor(v, sh, 32));
      float mnew = fmaxf(mi[r], v);
      float alpha = __expf(mi[r] - mnew);
      mi[r] = mnew;
      float rs = 0.f;
#pragma unroll
      for (int nt = 0; nt < 4; ++nt) {
        float p = __expf(s[nt][r] - mnew);
        s[nt][r] = p;
        rs += p;
      }
#pragma unroll
      for (int sh = 1; sh <= 8; sh <<= 1) rs += __shfl_xor(rs, sh, 32);
      li[r] = li[r] * alpha + rs;
#pragma unroll
      for (int d = 0; d < 4; ++d) o[d][r] *= alpha;
    }

    // P (C layout) -> LDS [q][kv] f16 (wave-private rows; no barrier needed)
#pragma unroll
    for (int nt = 0; nt < 4; ++nt)
#pragma unroll
      for (int r = 0; r < 8; ++r)
        Ps[(w * 16 + r + 8 * hf) * LDP + nt * 16 + nl] = (_Float16)s[nt][r];

    // O += P * V (contraction over kv=64; Vt is [n=d][k=kv])
#pragma unroll
    for (int ks = 0; ks < 2; ++ks) {
      v16h pf = load_afrag(Ps, w * 16 + nl, LDP, ks * 32, lane);
#pragma unroll
      for (int dt = 0; dt < 4; ++dt) {
        v16h vf = load_bfrag(Vt, dt * 16 + nl, LDV, ks * 32, lane);
        o[dt] = wmma_f16(pf, vf, o[dt]);
      }
    }
  }

  // Normalize and write context in (b, s, h*64+d) layout (feeds output GEMM)
#pragma unroll
  for (int r = 0; r < 8; ++r) {
    float inv = 1.0f / li[r];
    int m_g = qbase + w * 16 + r + 8 * hf;
    _Float16* dst = ctx + (size_t)(b * SEQ + m_g) * DMODEL + h * HDIM;
#pragma unroll
    for (int dt = 0; dt < 4; ++dt)
      dst[dt * 16 + nl] = (_Float16)(o[dt][r] * inv);
  }
}

// ---------------------------------------------------------------------------
extern "C" void kernel_launch(void* const* d_in, const int* in_sizes, int n_in,
                              void* d_out, int out_size, void* d_ws,
                              size_t ws_size, hipStream_t stream) {
  (void)in_sizes; (void)n_in; (void)out_size; (void)ws_size;
  const float* hidden = (const float*)d_in[0];
  const int*   amask  = (const int*)d_in[1];
  const float* Wq = (const float*)d_in[2];
  const float* bq = (const float*)d_in[3];
  const float* Wk = (const float*)d_in[4];
  const float* bk = (const float*)d_in[5];
  const float* Wv = (const float*)d_in[6];
  const float* bv = (const float*)d_in[7];
  const float* Wo = (const float*)d_in[8];
  const float* bo = (const float*)d_in[9];

  const size_t nX = (size_t)BATCH * SEQ * DMODEL;  // 8388608
  const size_t nW = (size_t)DMODEL * DMODEL;       // 4194304
  _Float16* Xh  = (_Float16*)d_ws;
  _Float16* Wqh = Xh + nX;
  _Float16* Wkh = Wqh + nW;
  _Float16* Wvh = Wkh + nW;
  _Float16* Woh = Wvh + nW;
  _Float16* Qh  = Woh + nW;
  _Float16* Kh  = Qh + nX;
  _Float16* Vh  = Kh + nX;
  _Float16* Ctx = Vh + nX;  // total ~112 MiB of f16 workspace

  cvt_f32_f16<<<(int)(nX / 1024), 256, 0, stream>>>(hidden, Xh, (int)(nX / 4));
  cvt_f32_f16<<<(int)(nW / 1024), 256, 0, stream>>>(Wq, Wqh, (int)(nW / 4));
  cvt_f32_f16<<<(int)(nW / 1024), 256, 0, stream>>>(Wk, Wkh, (int)(nW / 4));
  cvt_f32_f16<<<(int)(nW / 1024), 256, 0, stream>>>(Wv, Wvh, (int)(nW / 4));
  cvt_f32_f16<<<(int)(nW / 1024), 256, 0, stream>>>(Wo, Woh, (int)(nW / 4));

  dim3 ggrid(DMODEL / 128, (BATCH * SEQ) / 128);  // (16, 32)
  gemm128<false><<<ggrid, 256, 0, stream>>>(Xh, Wqh, bq, Qh);
  gemm128<false><<<ggrid, 256, 0, stream>>>(Xh, Wkh, bk, Kh);
  gemm128<false><<<ggrid, 256, 0, stream>>>(Xh, Wvh, bv, Vh);

  dim3 fgrid(SEQ / 128, NHEAD, BATCH);  // (16, 32, 2)
  flash64<<<fgrid, 256, 0, stream>>>(Qh, Kh, Vh, amask, Ctx);

  gemm128<true><<<ggrid, 256, 0, stream>>>(Ctx, Woh, bo, (float*)d_out);
}
